// LinearAttention_41798621725051
// MI455X (gfx1250) — compile-verified, matrix operand order checked by reading, not run on hardware
//
#include <hip/hip_runtime.h>

typedef __attribute__((ext_vector_type(16))) __bf16 v16bf;
typedef __attribute__((ext_vector_type(4)))  __bf16 v4bf;
typedef __attribute__((ext_vector_type(8)))  float  v8f;
typedef __attribute__((ext_vector_type(4)))  unsigned int u32x4;
typedef __attribute__((ext_vector_type(8)))  int i32x8;
typedef __attribute__((ext_vector_type(4)))  int i32x4;

#define SEQ    4096
#define BATCH  4
#define DMODEL 1024
#define HEADS  16
#define DHEAD  64
#define ROWS   (SEQ * BATCH)  // 16384
#define KTILE  32
#define MT     64             // rows per workgroup (16 s-groups x 4 b)
#define NT     64             // cols per workgroup (= one head)

#if defined(__has_builtin)
#if __has_builtin(__builtin_amdgcn_tensor_load_to_lds) && __has_builtin(__builtin_amdgcn_s_wait_tensorcnt)
#define USE_TDM 1
#endif
#endif
#ifndef USE_TDM
#define USE_TDM 0
#endif

#if USE_TDM
// Issue one TDM 2-D tile load: 64 rows x 32 f32, row stride 1024 elements,
// into LDS at lds_off. Descriptor per CDNA5 ISA §8.3/8.4.
// 6-arg builtin form: (g0, g1, g2, g3, g4, cpol).
__device__ __forceinline__ void tdm_load_tile_f32(unsigned lds_off, const float* gptr)
{
    unsigned long long ga = (unsigned long long)(uintptr_t)gptr;
    u32x4 g0;
    g0.x = 1u;                                                  // count=1, user mode
    g0.y = lds_off;                                             // LDS byte address
    g0.z = (unsigned)(ga & 0xFFFFFFFFull);                      // global_addr[31:0]
    g0.w = (unsigned)((ga >> 32) & 0x01FFFFFFull) | (2u << 30); // addr[56:32] | type=2
    i32x8 g1;
    g1[0] = (int)(2u << 16);           // workgroup_mask=0, data_size=2 (4 bytes)
    g1[1] = (int)(1024u << 16);        // tensor_dim0 = 1024 (bits 79:48, low half)
    g1[2] = 0;                         // tensor_dim0 hi16 = 0, tensor_dim1 lo16 = 0
    g1[3] = (int)(16u | (32u << 16));  // tensor_dim1 = 1<<20 (hi16=16), tile_dim0 = 32
    g1[4] = 64;                        // tile_dim1 = 64 rows, tile_dim2 = 0
    g1[5] = 1024;                      // tensor_dim0_stride = 1024 elements (lo32)
    g1[6] = 0;                         // stride hi, tensor_dim1_stride lo
    g1[7] = 0;
    i32x4 z4 = {0, 0, 0, 0};
    i32x8 z8 = {0, 0, 0, 0, 0, 0, 0, 0};
    __builtin_amdgcn_tensor_load_to_lds(g0, g1, z4, z4, z8, 0);
}
#endif

__global__ __launch_bounds__(256)
void la_fused_kernel(const float* __restrict__ q,
                     const float* __restrict__ k,
                     const float* __restrict__ v,
                     const float* __restrict__ Wq,
                     const float* __restrict__ Wk,
                     const float* __restrict__ Wv,
                     float* __restrict__ out)
{
#if USE_TDM
    __shared__ float sF[6][MT][KTILE];     // 48 KB f32 staging (TDM destination)
#endif
    __shared__ __bf16 sB16[6][MT][KTILE];  // 24 KB bf16 operands: 0-2 act, 3-5 weights
    __shared__ float  sR[3][MT][NT];       // 48 KB results: exp(qWq), exp(kWk), vWv
    __shared__ float  sQK[MT];

    const int tid  = threadIdx.x;
    const int lane = tid & 31;
    const int wave = tid >> 5;
    const int h    = blockIdx.x;           // head fastest-varying -> L2 reuse of q/k/v rows
    const int row0 = blockIdx.y * MT;      // global row = s*BATCH + b
    const int n0   = h * DHEAD;

    // wave tiling: mb = 16-row block (0..3), nb = 32-col block (0..1)
    const int mb = wave & 3;
    const int nb = wave >> 2;
    // 16-bit A/B operand layout: lane (L&15) = m/n; hi = L>>4 selects K-halves
    const int lsel = lane & 15;
    const int hi   = lane >> 4;

    v8f acc[3][2];
#pragma unroll
    for (int g = 0; g < 3; ++g)
#pragma unroll
        for (int t = 0; t < 2; ++t)
            acc[g][t] = (v8f)0.0f;

    for (int kk = 0; kk < DMODEL; kk += KTILE) {
#if USE_TDM
        // ---- async DMA: 6 f32 tiles global -> LDS via Tensor Data Mover ----
        if (wave == 0) {
            tdm_load_tile_f32((unsigned)(uintptr_t)&sF[0][0][0], q  + (size_t)row0 * DMODEL + kk);
            tdm_load_tile_f32((unsigned)(uintptr_t)&sF[1][0][0], k  + (size_t)row0 * DMODEL + kk);
            tdm_load_tile_f32((unsigned)(uintptr_t)&sF[2][0][0], v  + (size_t)row0 * DMODEL + kk);
            tdm_load_tile_f32((unsigned)(uintptr_t)&sF[3][0][0], Wq + (size_t)n0   * DMODEL + kk);
            tdm_load_tile_f32((unsigned)(uintptr_t)&sF[4][0][0], Wk + (size_t)n0   * DMODEL + kk);
            tdm_load_tile_f32((unsigned)(uintptr_t)&sF[5][0][0], Wv + (size_t)n0   * DMODEL + kk);
            __builtin_amdgcn_s_wait_tensorcnt(0);
        }
        __syncthreads();

        // ---- convert f32 LDS -> bf16 LDS (once per element) ----
#pragma unroll
        for (int it = 0; it < 12; ++it) {
            int t    = tid + it * 256;     // 0..3071 float4-quads
            int tile = t >> 9;             // constant per unrolled it
            int rem  = t & 511;
            int i    = rem >> 3;
            int qd   = rem & 7;
            float4 f = *(const float4*)&sF[tile][i][qd * 4];
            v4bf b4;
            b4.x = (__bf16)f.x; b4.y = (__bf16)f.y;
            b4.z = (__bf16)f.z; b4.w = (__bf16)f.w;
            *(v4bf*)&sB16[tile][i][qd * 4] = b4;
        }
        __syncthreads();
#else
        // ---- fallback: direct global_load staging with on-the-fly conversion ----
        __syncthreads();
#pragma unroll
        for (int it = 0; it < 12; ++it) {
            int t    = tid + it * 256;
            int tile = t >> 9;             // constant per unrolled it
            int rem  = t & 511;
            int i    = rem >> 3;
            int qd   = rem & 7;
            const float* src;
            if      (tile == 0) src = q  + (size_t)(row0 + i) * DMODEL + kk + qd * 4;
            else if (tile == 1) src = k  + (size_t)(row0 + i) * DMODEL + kk + qd * 4;
            else if (tile == 2) src = v  + (size_t)(row0 + i) * DMODEL + kk + qd * 4;
            else if (tile == 3) src = Wq + (size_t)(n0  + i) * DMODEL + kk + qd * 4;
            else if (tile == 4) src = Wk + (size_t)(n0  + i) * DMODEL + kk + qd * 4;
            else                src = Wv + (size_t)(n0  + i) * DMODEL + kk + qd * 4;
            float4 f = *(const float4*)src;
            v4bf b4;
            b4.x = (__bf16)f.x; b4.y = (__bf16)f.y;
            b4.z = (__bf16)f.z; b4.w = (__bf16)f.w;
            *(v4bf*)&sB16[tile][i][qd * 4] = b4;
        }
        __syncthreads();
#endif

        // ---- 3 GEMM tiles via WMMA bf16 ----
#pragma unroll
        for (int g = 0; g < 3; ++g) {
            v16bf a;
            {
                const __bf16* base = &sB16[g][mb * 16 + lsel][8 * hi];
                ((float4*)&a)[0] = *(const float4*)(base);        // K = 8*hi .. +7
                ((float4*)&a)[1] = *(const float4*)(base + 16);   // K = 16+8*hi .. +7
            }
#pragma unroll
            for (int t = 0; t < 2; ++t) {
                v16bf bm;
                const __bf16* base = &sB16[3 + g][nb * 32 + t * 16 + lsel][8 * hi];
                ((float4*)&bm)[0] = *(const float4*)(base);
                ((float4*)&bm)[1] = *(const float4*)(base + 16);
                acc[g][t] = __builtin_amdgcn_wmma_f32_16x16x32_bf16(
                    false, a, false, bm, (short)0, acc[g][t], false, false);
            }
        }
    }
    __syncthreads();

    // ---- spill accumulators to LDS result tiles (exp fused for q,k) ----
    // C layout: lane holds column lsel; VGPR j -> row j + 8*hi
#pragma unroll
    for (int g = 0; g < 3; ++g) {
#pragma unroll
        for (int t = 0; t < 2; ++t) {
            int col = nb * 32 + t * 16 + lsel;
#pragma unroll
            for (int j = 0; j < 8; ++j) {
                int r = mb * 16 + hi * 8 + j;
                float val = acc[g][t][j];
                if (g < 2) val = __expf(val);
                sR[g][r][col] = val;
            }
        }
    }
    __syncthreads();

    // ---- per-row qk = sum over head dims of exp(q)*exp(k) ----
    if (tid < MT) {
        float s = 0.0f;
#pragma unroll
        for (int j = 0; j < NT; ++j)
            s += sR[0][tid][j] * sR[1][tid][j];
        sQK[tid] = s;
    }
    __syncthreads();

    // ---- out = (qk*v)/qk with cumsum over b (4 consecutive rows) ----
#pragma unroll
    for (int it = 0; it < 4; ++it) {
        int idx = tid + it * 256;          // 16 s-groups x 64 dims
        int sl  = idx >> 6;
        int d   = idx & 63;
        float run = 0.0f;
#pragma unroll
        for (int b = 0; b < BATCH; ++b) {
            int r = sl * 4 + b;
            float qk = sQK[r];
            run += (qk * sR[2][r][d]) / qk;
            out[(size_t)(row0 + r) * DMODEL + n0 + d] = run;
        }
    }
}

extern "C" void kernel_launch(void* const* d_in, const int* in_sizes, int n_in,
                              void* d_out, int out_size, void* d_ws, size_t ws_size,
                              hipStream_t stream) {
    const float* q  = (const float*)d_in[0];
    const float* k  = (const float*)d_in[1];
    const float* v  = (const float*)d_in[2];
    const float* Wq = (const float*)d_in[3];
    const float* Wk = (const float*)d_in[4];
    const float* Wv = (const float*)d_in[5];
    float* out = (float*)d_out;

    dim3 grid(HEADS, ROWS / MT);   // head fastest -> activation reuse in L2
    la_fused_kernel<<<grid, dim3(256), 0, stream>>>(q, k, v, Wq, Wk, Wv, out);
}
